// Head_19550691131727
// MI455X (gfx1250) — matrix operand order repaired and checked by reading, not compiled, not run on hardware
//
#include <hip/hip_runtime.h>
#include <hip/hip_bf16.h>
#include <math.h>

// Problem constants (compile-time, from the reference).
#define BB   4
#define TT   2048
#define CC   1024
#define DD   64
#define ROWS (BB * TT)          // 8192 total sequence rows

typedef __attribute__((ext_vector_type(16))) _Float16 v16h;
typedef __attribute__((ext_vector_type(8)))  _Float16 v8h;
typedef __attribute__((ext_vector_type(8)))  float    v8f;
typedef __attribute__((ext_vector_type(4)))  float    v4f;

// Load a 16-half A-operand slice: two contiguous 8-half (16 B) runs at p and p+8.
// Contiguous-K A operand (B packed in the same order): lane row = lane%16,
// K offsets = kbase + {0..15} with kbase = (lane/16)*16.
__device__ __forceinline__ v16h load_a16(const _Float16* __restrict__ p) {
    v8h r0 = *(const v8h*)(p);
    v8h r1 = *(const v8h*)(p + 8);
    v16h a;
#pragma unroll
    for (int i = 0; i < 8; ++i) { a[i] = r0[i]; a[i + 8] = r1[i]; }
    return a;
}

// ---------------------------------------------------------------------------
// Kernel 0: pack Wq/Wk/Wv (f32 [1024][64]) into the exact WMMA B-operand
// register image, f16.  Layout: wpack[(mat*128 + kb*4 + ct)*32 + lane][16].
// B image in contiguous K order (as A): col = lane%16, K = (lane/16)*16 + h.
// ---------------------------------------------------------------------------
__global__ void pack_w(const float* __restrict__ Wq,
                       const float* __restrict__ Wk,
                       const float* __restrict__ Wv,
                       _Float16* __restrict__ wpack) {
    int bx   = blockIdx.x;             // 0..383 = mat*128 + kb*4 + ct
    int mat  = bx >> 7;
    int kb   = (bx >> 2) & 31;
    int ct   = bx & 3;
    const float* W = (mat == 0) ? Wq : (mat == 1) ? Wk : Wv;
    int lane = threadIdx.x;
    int col  = ct * 16 + (lane & 15);
    int krow = kb * 32 + (lane >> 4) * 16;
    v16h pk;
#pragma unroll
    for (int h = 0; h < 16; ++h)
        pk[h] = (_Float16)W[(size_t)(krow + h) * DD + col];
    *(v16h*)(wpack + ((size_t)bx * 32 + lane) * 16) = pk;
}

// ---------------------------------------------------------------------------
// Kernel 1: QKV projection.  grid = 512 row-tiles, block = 128 (4 waves);
// wave w owns output column-tile w (16 of the 64 head dims) for q, k, v.
// q stored row-major f16 with 1/sqrt(C)=1/32 folded in; k row-major f16;
// v stored TRANSPOSED per batch: vT[b][d][t]  (for contiguous PV A-loads).
// ---------------------------------------------------------------------------
__global__ void qkv_proj(const float* __restrict__ x,
                         const _Float16* __restrict__ wpack,
                         _Float16* __restrict__ qh,
                         _Float16* __restrict__ kh,
                         _Float16* __restrict__ vT) {
    int rt      = blockIdx.x;               // 0..511
    int w       = threadIdx.x >> 5;         // colTile 0..3
    int lane    = threadIdx.x & 31;
    int hh      = lane >> 4;                // half of the wave
    int rowBase = rt * 16;
    int row     = rowBase + (lane & 15);
    int kbase   = hh * 16;
    const float* xr = x + (size_t)row * CC;

    v8f aq = {}, ak = {}, av = {};
    for (int kb = 0; kb < 32; ++kb) {
        // A tile (x, f32 -> f16): four 16 B loads per lane (16 contiguous floats).
        const float* p = xr + kb * 32 + kbase;
        v4f f0 = *(const v4f*)(p);
        v4f f1 = *(const v4f*)(p + 4);
        v4f f2 = *(const v4f*)(p + 8);
        v4f f3 = *(const v4f*)(p + 12);
        v16h a;
#pragma unroll
        for (int i = 0; i < 4; ++i) {
            a[i]      = (_Float16)f0[i];
            a[i + 4]  = (_Float16)f1[i];
            a[i + 8]  = (_Float16)f2[i];
            a[i + 12] = (_Float16)f3[i];
        }
        // B tiles: pre-packed, one 32 B contiguous load per lane per matrix.
        const _Float16* wp = wpack + ((size_t)(kb * 4 + w) * 32 + lane) * 16;
        v16h bq = *(const v16h*)(wp);
        v16h bk = *(const v16h*)(wp + 65536);    // mat stride = 128*32*16
        v16h bv = *(const v16h*)(wp + 131072);
        aq = __builtin_amdgcn_wmma_f32_16x16x32_f16(false, a, false, bq, (short)0, aq, false, false);
        ak = __builtin_amdgcn_wmma_f32_16x16x32_f16(false, a, false, bk, (short)0, ak, false, false);
        av = __builtin_amdgcn_wmma_f32_16x16x32_f16(false, a, false, bv, (short)0, av, false, false);
    }

    // C/D layout: n = lane%16, row m = r + 8*(lane/16).
    int n       = lane & 15;
    int colBase = w * 16;
    int b       = rowBase / TT;
    int tBase   = rowBase % TT;
#pragma unroll
    for (int r = 0; r < 8; ++r) {
        int m = r + 8 * hh;
        size_t ro = (size_t)(rowBase + m) * DD + colBase + n;
        qh[ro] = (_Float16)(aq[r] * 0.03125f);   // fold 1/sqrt(1024)
        kh[ro] = (_Float16)ak[r];
        vT[(size_t)b * (DD * TT) + (size_t)(colBase + n) * TT + tBase + m] = (_Float16)av[r];
    }
}

// ---------------------------------------------------------------------------
// Kernel 2: causal flash attention, one wave per (batch, 16-query tile).
// Computes S^T = K·Q^T (query = lane%16 -> softmax reductions are in-lane
// plus one shfl_xor(16)), then out^T = V^T·P^T where P^T's B-operand is
// built from S^T's D-registers with sixteen half-swap shuffles (no LDS).
// ---------------------------------------------------------------------------
__global__ void flash_attn(const _Float16* __restrict__ qh,
                           const _Float16* __restrict__ kh,
                           const _Float16* __restrict__ vT,
                           float* __restrict__ out) {
    int bidx  = blockIdx.x;            // 0..511 = b*128 + qi
    int b     = bidx >> 7;
    int qi    = bidx & 127;
    int lane  = threadIdx.x;
    int hh    = lane >> 4;
    int nq    = lane & 15;
    bool low  = (lane < 16);
    int qSeq  = qi * 16 + nq;          // query position (causal mask)
    int qRowG = b * TT + qSeq;
    int kbA   = hh * 16;               // A-layout K-offset base

    // Q^T B-tiles (contraction over D=64 -> two 32-deep tiles), loaded once.
    const _Float16* qp = qh + (size_t)qRowG * DD + hh * 16;
    v16h bQ0 = *(const v16h*)(qp);
    v16h bQ1 = *(const v16h*)(qp + 32);

    v8f o0 = {}, o1 = {}, o2 = {}, o3 = {};      // out^T accum, d-tiles 0..3
    float m_run = -__builtin_inff();
    float l_run = 0.0f;

    int nkb = (qi >> 1) + 1;                     // causal 32-key blocks
    const _Float16* kbatch = kh + (size_t)b * TT * DD;
    const _Float16* vbatch = vT + (size_t)b * (DD * TT);

    for (int kb = 0; kb < nkb; ++kb) {
        int keyBase = kb * 32;
        if (kb + 1 < nkb)   // prefetch next K block (gfx1250 global_prefetch)
            __builtin_prefetch(kbatch + (size_t)(keyBase + 32 + nq) * DD, 0, 1);

        // ---- S^T tiles: M = keys (two 16-row tiles), N = queries ----
        const _Float16* kp0 = kbatch + (size_t)(keyBase + nq) * DD;
        const _Float16* kp1 = kbatch + (size_t)(keyBase + 16 + nq) * DD;
        v8f st0 = {}, st1 = {};
        st0 = __builtin_amdgcn_wmma_f32_16x16x32_f16(false, load_a16(kp0 + kbA),      false, bQ0, (short)0, st0, false, false);
        st0 = __builtin_amdgcn_wmma_f32_16x16x32_f16(false, load_a16(kp0 + 32 + kbA), false, bQ1, (short)0, st0, false, false);
        st1 = __builtin_amdgcn_wmma_f32_16x16x32_f16(false, load_a16(kp1 + kbA),      false, bQ0, (short)0, st1, false, false);
        st1 = __builtin_amdgcn_wmma_f32_16x16x32_f16(false, load_a16(kp1 + 32 + kbA), false, bQ1, (short)0, st1, false, false);

        float s0[8], s1[8];
#pragma unroll
        for (int r = 0; r < 8; ++r) { s0[r] = st0[r]; s1[r] = st1[r]; }

        // Causal mask only on the diagonal block.
        if (kb == nkb - 1) {
#pragma unroll
            for (int r = 0; r < 8; ++r) {
                int key0 = keyBase + r + 8 * hh;
                int key1 = key0 + 16;
                if (key0 > qSeq) s0[r] = -__builtin_inff();
                if (key1 > qSeq) s1[r] = -__builtin_inff();
            }
        }

        // Row max: in-lane over 16 values + one half-merge shuffle.
        float vmax = s0[0];
#pragma unroll
        for (int r = 1; r < 8; ++r) vmax = fmaxf(vmax, s0[r]);
#pragma unroll
        for (int r = 0; r < 8; ++r) vmax = fmaxf(vmax, s1[r]);
        vmax = fmaxf(vmax, __shfl_xor(vmax, 16));
        float m_new = fmaxf(m_run, vmax);
        float alpha = __expf(m_run - m_new);     // exp(-inf)=0 on first block

        float p0[8], p1[8], rs = 0.0f;
#pragma unroll
        for (int r = 0; r < 8; ++r) { p0[r] = __expf(s0[r] - m_new); rs += p0[r]; }
#pragma unroll
        for (int r = 0; r < 8; ++r) { p1[r] = __expf(s1[r] - m_new); rs += p1[r]; }
        rs += __shfl_xor(rs, 16);
        l_run = l_run * alpha + rs;
        m_run = m_new;

#pragma unroll
        for (int r = 0; r < 8; ++r) { o0[r] *= alpha; o1[r] *= alpha; o2[r] *= alpha; o3[r] *= alpha; }

        // Build P^T B-operand (32 keys x 16 queries) via half-swap shuffles.
        v16h bP;
#pragma unroll
        for (int h = 0; h < 8; ++h) {
            float other = __shfl_xor(p1[h], 16);
            bP[h] = (_Float16)(low ? p0[h] : other);
        }
#pragma unroll
        for (int h = 0; h < 8; ++h) {
            float other = __shfl_xor(p0[h], 16);
            bP[h + 8] = (_Float16)(low ? other : p1[h]);
        }

        // out^T += V^T · P^T  (A tiles from transposed V: contiguous runs).
        const _Float16* vp = vbatch + (size_t)nq * TT + keyBase + kbA;
        o0 = __builtin_amdgcn_wmma_f32_16x16x32_f16(false, load_a16(vp),           false, bP, (short)0, o0, false, false);
        o1 = __builtin_amdgcn_wmma_f32_16x16x32_f16(false, load_a16(vp + 16 * TT), false, bP, (short)0, o1, false, false);
        o2 = __builtin_amdgcn_wmma_f32_16x16x32_f16(false, load_a16(vp + 32 * TT), false, bP, (short)0, o2, false, false);
        o3 = __builtin_amdgcn_wmma_f32_16x16x32_f16(false, load_a16(vp + 48 * TT), false, bP, (short)0, o3, false, false);
    }

    // Finalize: divide by l, write out[b][t][d] (all stores land in one row).
    float inv = 1.0f / l_run;
    float* op = out + (size_t)qRowG * DD;
#pragma unroll
    for (int r = 0; r < 8; ++r) {
        int d = r + 8 * hh;
        op[d]      = o0[r] * inv;
        op[16 + d] = o1[r] * inv;
        op[32 + d] = o2[r] * inv;
        op[48 + d] = o3[r] * inv;
    }
}

// ---------------------------------------------------------------------------
extern "C" void kernel_launch(void* const* d_in, const int* in_sizes, int n_in,
                              void* d_out, int out_size, void* d_ws, size_t ws_size,
                              hipStream_t stream) {
    (void)in_sizes; (void)n_in; (void)out_size; (void)ws_size;
    const float* x  = (const float*)d_in[0];
    const float* Wq = (const float*)d_in[1];
    const float* Wk = (const float*)d_in[2];
    const float* Wv = (const float*)d_in[3];

    // Workspace layout (f16): q | k | vT | packed weights  (~3.4 MB total)
    _Float16* qh    = (_Float16*)d_ws;
    _Float16* kh    = qh + (size_t)ROWS * DD;
    _Float16* vT    = kh + (size_t)ROWS * DD;
    _Float16* wpack = vT + (size_t)ROWS * DD;

    pack_w    <<<384, 32,  0, stream>>>(Wq, Wk, Wv, wpack);
    qkv_proj  <<<512, 128, 0, stream>>>(x, wpack, qh, kh, vT);
    flash_attn<<<512, 32,  0, stream>>>(qh, kh, vT, (float*)d_out);
}
